// CrystalGNN_22041772163193
// MI455X (gfx1250) — matrix operand order repaired and checked by reading, not compiled, not run on hardware
//
#include <hip/hip_runtime.h>
#include <hip/hip_bf16.h>

#define N_NODES   100000
#define N_FEAT    92
#define HID       128
#define N_EDGES   1600000
#define N_GRAPHS  2048
#define BN_EPS    1e-5f

typedef __attribute__((ext_vector_type(8)))  float          v8f;
typedef __attribute__((ext_vector_type(8)))  unsigned short v8u;
typedef __attribute__((ext_vector_type(16))) unsigned short v16u;
typedef __attribute__((ext_vector_type(16))) __bf16         v16bf;

__device__ __forceinline__ unsigned short bf16bits(float f) {
    unsigned u = __builtin_bit_cast(unsigned, f);
    return (unsigned short)((u + 0x7FFFu + ((u >> 16) & 1u)) >> 16);
}

__device__ __forceinline__ void atomAddF(float* p, float v) {
    __hip_atomic_fetch_add(p, v, __ATOMIC_RELAXED, __HIP_MEMORY_SCOPE_AGENT);
}

// ---------------------------------------------------------------- utilities
__global__ void zero_f32(float* __restrict__ p, int n) {
    int i = blockIdx.x * 256 + threadIdx.x;
    if (i < n) p[i] = 0.0f;
}

__global__ void deg_kernel(const int* __restrict__ dst, float* __restrict__ deg, int E) {
    int e = blockIdx.x * 256 + threadIdx.x;
    if (e < E) atomAddF(&deg[dst[e]], 1.0f);
}

__global__ void dinv_kernel(const float* __restrict__ deg, float* __restrict__ dinv, int n) {
    int i = blockIdx.x * 256 + threadIdx.x;
    if (i < n) {
        float d = deg[i];
        dinv[i] = (d > 0.0f) ? rsqrtf(fmaxf(d, 1.0f)) : 0.0f;
    }
}

__global__ void we_kernel(const int* __restrict__ src, const int* __restrict__ dst,
                          const float* __restrict__ dinv, float* __restrict__ we, int E) {
    int e = blockIdx.x * 256 + threadIdx.x;
    if (e < E) we[e] = dinv[src[e]] * dinv[dst[e]];
}

// W [K, Hout] f32 row-major  ->  Bt [Hout, Kpad] bf16 (transposed, zero-padded K)
__global__ void convert_wT(const float* __restrict__ W, unsigned short* __restrict__ Bt,
                           int K, int Kpad, int Hout) {
    int idx = blockIdx.x * 256 + threadIdx.x;
    if (idx >= Hout * Kpad) return;
    int n = idx / Kpad, k = idx - n * Kpad;
    float v = (k < K) ? W[(size_t)k * Hout + n] : 0.0f;
    Bt[idx] = bf16bits(v);
}

// ---------------------------------------------------------------- WMMA GEMM
// C[M,128] (=/+=) A[M,Kreal] @ B[Kreal,128], B pre-transposed bf16 [128][KPAD].
// One block = 4 waves; block computes rows [32*bid, 32*bid+32) x all 128 cols.
// A tile converted to bf16 ONCE into LDS; each wave: 2 M-tiles x 2 N-tiles.
template <int KPAD>
__global__ __launch_bounds__(128) void gemm_tile32(
    const float* __restrict__ A, int lda, int Kreal,
    const unsigned short* __restrict__ Bt,
    float* __restrict__ C, int M, int initC)
{
    constexpr int STRIDE = KPAD + 8;                 // u16 units, 16B-aligned rows
    __shared__ unsigned short As[32 * STRIDE];
    unsigned int* As32 = (unsigned int*)As;
    const int tid  = threadIdx.x;
    const int row0 = blockIdx.x * 32;

    // stage A tile as bf16 (convert once per element per block)
    for (int p = tid; p < 32 * (KPAD / 2); p += 128) {
        int r  = p / (KPAD / 2);
        int kp = p - r * (KPAD / 2);
        int k  = kp * 2;
        int gr = row0 + r;
        float f0 = 0.0f, f1 = 0.0f;
        if (gr < M) {
            if (k < Kreal)     f0 = A[(size_t)gr * lda + k];
            if (k + 1 < Kreal) f1 = A[(size_t)gr * lda + k + 1];
        }
        As32[r * (STRIDE / 2) + kp] =
            (unsigned)bf16bits(f0) | ((unsigned)bf16bits(f1) << 16);
    }
    __syncthreads();

    const int lane = tid & 31;
    const int wave = tid >> 5;
    const int colbase = wave * 32;                   // wave covers 2 N-tiles of 16
    const int nlo = lane & 15;
    const int hi  = lane >> 4;
    const int m   = lane & 15;

    v8f c00 = {}, c01 = {}, c10 = {}, c11 = {};
    if (!initC) {
#pragma unroll
        for (int v = 0; v < 8; ++v) {
            int g0 = row0 + hi * 8 + v;
            int g1 = row0 + 16 + hi * 8 + v;
            c00[v] = C[(size_t)g0 * HID + colbase + nlo];
            c01[v] = C[(size_t)g0 * HID + colbase + 16 + nlo];
            c10[v] = C[(size_t)g1 * HID + colbase + nlo];
            c11[v] = C[(size_t)g1 * HID + colbase + 16 + nlo];
        }
    }

#pragma unroll
    for (int kt = 0; kt < KPAD / 32; ++kt) {
        // A fragments (16x32 bf16): lanes 0-15 K=0..7/16..23, lanes 16-31 K=8..15/24..31
        const unsigned short* a0p = &As[m * STRIDE + kt * 32 + hi * 8];
        const unsigned short* a1p = &As[(16 + m) * STRIDE + kt * 32 + hi * 8];
        v8u a0lo = *(const v8u*)a0p;
        v8u a0hi = *(const v8u*)(a0p + 16);
        v8u a1lo = *(const v8u*)a1p;
        v8u a1hi = *(const v8u*)(a1p + 16);
        v16bf a0 = __builtin_bit_cast(v16bf, __builtin_shufflevector(
            a0lo, a0hi, 0,1,2,3,4,5,6,7,8,9,10,11,12,13,14,15));
        v16bf a1 = __builtin_bit_cast(v16bf, __builtin_shufflevector(
            a1lo, a1hi, 0,1,2,3,4,5,6,7,8,9,10,11,12,13,14,15));

        // B fragments (32x16): lane column = lane%16, lanes 0-15 K=0..7/16..23, lanes 16-31 K=8..15/24..31
        const unsigned short* bp0 = Bt + (size_t)(colbase + nlo) * KPAD + kt * 32 + hi * 8;
        const unsigned short* bp1 = Bt + (size_t)(colbase + 16 + nlo) * KPAD + kt * 32 + hi * 8;
        v16bf b0 = __builtin_bit_cast(v16bf, __builtin_shufflevector(
            *(const v8u*)bp0, *(const v8u*)(bp0 + 16), 0,1,2,3,4,5,6,7,8,9,10,11,12,13,14,15));
        v16bf b1 = __builtin_bit_cast(v16bf, __builtin_shufflevector(
            *(const v8u*)bp1, *(const v8u*)(bp1 + 16), 0,1,2,3,4,5,6,7,8,9,10,11,12,13,14,15));

        c00 = __builtin_amdgcn_wmma_f32_16x16x32_bf16(false, a0, false, b0, (short)0, c00, false, false);
        c01 = __builtin_amdgcn_wmma_f32_16x16x32_bf16(false, a0, false, b1, (short)0, c01, false, false);
        c10 = __builtin_amdgcn_wmma_f32_16x16x32_bf16(false, a1, false, b0, (short)0, c10, false, false);
        c11 = __builtin_amdgcn_wmma_f32_16x16x32_bf16(false, a1, false, b1, (short)0, c11, false, false);
    }

#pragma unroll
    for (int v = 0; v < 8; ++v) {
        int g0 = row0 + hi * 8 + v;
        int g1 = row0 + 16 + hi * 8 + v;
        if (g0 < M) {
            C[(size_t)g0 * HID + colbase + nlo]      = c00[v];
            C[(size_t)g0 * HID + colbase + 16 + nlo] = c01[v];
        }
        if (g1 < M) {
            C[(size_t)g1 * HID + colbase + nlo]      = c10[v];
            C[(size_t)g1 * HID + colbase + 16 + nlo] = c11[v];
        }
    }
}

// ---------------------------------------------------------------- elementwise
__global__ void bias_relu(const float* __restrict__ acc, const float* __restrict__ b,
                          float* __restrict__ out, int n) {
    int i = blockIdx.x * 256 + threadIdx.x;
    if (i < n) out[i] = fmaxf(acc[i] + b[i & (HID - 1)], 0.0f);
}

// out[dst] += w_e * h[src]  (one thread per (edge, 4-channel group))
__global__ void prop_kernel(const float* __restrict__ hin, float* __restrict__ hout,
                            const int* __restrict__ src, const int* __restrict__ dst,
                            const float* __restrict__ we, int E) {
    int idx = blockIdx.x * 256 + threadIdx.x;
    if (idx >= E * 32) return;
    int e = idx >> 5;
    int q = idx & 31;
    float w = we[e];
    int s = src[e], d = dst[e];
    const float4 hv = *(const float4*)(hin + (size_t)s * HID + q * 4);
    float* o = hout + (size_t)d * HID + q * 4;
    atomAddF(o + 0, w * hv.x);
    atomAddF(o + 1, w * hv.y);
    atomAddF(o + 2, w * hv.z);
    atomAddF(o + 3, w * hv.w);
}

__global__ __launch_bounds__(128) void bn_stats(const float* __restrict__ acc,
                                                float* __restrict__ sum, float* __restrict__ sq, int N) {
    int c = threadIdx.x;
    int r0 = blockIdx.x * 256;
    int rend = min(r0 + 256, N);
    float s = 0.0f, q = 0.0f;
    for (int r = r0; r < rend; ++r) {
        float v = acc[(size_t)r * HID + c];
        s += v; q += v * v;
    }
    atomAddF(&sum[c], s);
    atomAddF(&sq[c],  q);
}

__global__ void bn_finalize(const float* __restrict__ sum, const float* __restrict__ sq,
                            const float* __restrict__ g, const float* __restrict__ b,
                            float* __restrict__ scale, float* __restrict__ shift, int N) {
    int c = threadIdx.x;
    float invN = 1.0f / (float)N;
    float mean = sum[c] * invN;
    float var  = sq[c] * invN - mean * mean;
    float sc = g[c] * rsqrtf(var + BN_EPS);
    scale[c] = sc;
    shift[c] = b[c] - mean * sc;
}

__global__ void bn_apply(const float* __restrict__ acc, const float* __restrict__ scale,
                         const float* __restrict__ shift, float* __restrict__ out, int n) {
    int i = blockIdx.x * 256 + threadIdx.x;
    if (i < n) {
        int c = i & (HID - 1);
        out[i] = fmaxf(acc[i] * scale[c] + shift[c], 0.0f);
    }
}

__global__ void lat_enc(const float* __restrict__ lf, const float* __restrict__ W,
                        const float* __restrict__ b, float* __restrict__ lat) {
    int g = blockIdx.x, c = threadIdx.x;
    float a = b[c];
#pragma unroll
    for (int k = 0; k < 9; ++k) a += lf[g * 9 + k] * W[k * HID + c];
    lat[(size_t)g * HID + c] = fmaxf(a, 0.0f);
}

__global__ void pool_sum(const float* __restrict__ h, const int* __restrict__ batch,
                         float* __restrict__ xsum, int n) {
    int i = blockIdx.x * 256 + threadIdx.x;
    if (i < n) {
        int node = i >> 7, c = i & (HID - 1);
        atomAddF(&xsum[(size_t)batch[node] * HID + c], h[i]);
    }
}

__global__ void pool_cnt(const int* __restrict__ batch, float* __restrict__ counts, int N) {
    int i = blockIdx.x * 256 + threadIdx.x;
    if (i < N) atomAddF(&counts[batch[i]], 1.0f);
}

__global__ __launch_bounds__(128) void head_kernel(
    const float* __restrict__ xsum, const float* __restrict__ counts, const float* __restrict__ lat,
    const float* __restrict__ fc1w, const float* __restrict__ fc1b,
    const float* __restrict__ fc2w, const float* __restrict__ fc2b,
    const float* __restrict__ fc3w, const float* __restrict__ fc3b,
    float* __restrict__ out)
{
    __shared__ float z[3 * HID];
    __shared__ float t1[HID];
    __shared__ float t2[64];
    int g = blockIdx.x, c = threadIdx.x;
    float cnt = fmaxf(counts[g], 1.0f);
    float s = xsum[(size_t)g * HID + c];
    z[c]           = s / cnt;           // mean
    z[HID + c]     = s;                 // sum
    z[2 * HID + c] = lat[(size_t)g * HID + c];
    __syncthreads();
    float a = fc1b[c];
    for (int k = 0; k < 3 * HID; ++k) a += z[k] * fc1w[k * HID + c];
    t1[c] = fmaxf(a, 0.0f);
    __syncthreads();
    if (c < 64) {
        float a2 = fc2b[c];
        for (int k = 0; k < HID; ++k) a2 += t1[k] * fc2w[k * 64 + c];
        t2[c] = fmaxf(a2, 0.0f) * fc3w[c];
    }
    __syncthreads();
    if (c == 0) {
        float r = fc3b[0];
        for (int k = 0; k < 64; ++k) r += t2[k];
        out[g] = r;
    }
}

// ---------------------------------------------------------------- launch
static inline int gdiv(long long n) { return (int)((n + 255) / 256); }

extern "C" void kernel_launch(void* const* d_in, const int* in_sizes, int n_in,
                              void* d_out, int out_size, void* d_ws, size_t ws_size,
                              hipStream_t stream)
{
    (void)in_sizes; (void)n_in; (void)out_size; (void)ws_size;
    const int N = N_NODES, E = N_EDGES, G = N_GRAPHS;

    const float* x      = (const float*)d_in[0];
    const int*   ei     = (const int*)d_in[1];
    const int*   batch  = (const int*)d_in[2];
    const float* latf   = (const float*)d_in[3];
    const float* node_w = (const float*)d_in[4];
    const float* node_b = (const float*)d_in[5];
    const float* lat_w  = (const float*)d_in[6];
    const float* lat_b  = (const float*)d_in[7];
    const float* fc1w   = (const float*)d_in[8];
    const float* fc1b   = (const float*)d_in[9];
    const float* fc2w   = (const float*)d_in[10];
    const float* fc2b   = (const float*)d_in[11];
    const float* fc3w   = (const float*)d_in[12];
    const float* fc3b   = (const float*)d_in[13];
    const float* convw[3] = {(const float*)d_in[14], (const float*)d_in[18], (const float*)d_in[22]};
    const float* bng[3]   = {(const float*)d_in[16], (const float*)d_in[20], (const float*)d_in[24]};
    const float* bnb[3]   = {(const float*)d_in[17], (const float*)d_in[21], (const float*)d_in[25]};
    const int* src = ei;
    const int* dst = ei + E;

    char* wsb = (char*)d_ws;
    size_t off = 0;
    auto alloc = [&](size_t bytes) -> char* {
        char* p = wsb + off;
        off = (off + bytes + 255) & ~(size_t)255;
        return p;
    };
    float* deg    = (float*)alloc((size_t)N * 4);
    float* dinv   = (float*)alloc((size_t)N * 4);
    float* we     = (float*)alloc((size_t)E * 4);
    float* h      = (float*)alloc((size_t)N * HID * 4);
    float* pA     = (float*)alloc((size_t)N * HID * 4);
    float* acc    = (float*)alloc((size_t)N * HID * 4);
    float* bnbuf  = (float*)alloc(512 * 4);     // sum | sq | scale | shift
    float* lat    = (float*)alloc((size_t)G * HID * 4);
    float* xsum   = (float*)alloc((size_t)G * HID * 4);
    float* counts = (float*)alloc((size_t)G * 4);
    unsigned short* nodeBt = (unsigned short*)alloc((size_t)HID * 96 * 2);
    unsigned short* convBt = (unsigned short*)alloc((size_t)12 * HID * HID * 2);

    const int NH = N * HID;

    // edge normalization weights
    zero_f32<<<gdiv(N), 256, 0, stream>>>(deg, N);
    deg_kernel<<<gdiv(E), 256, 0, stream>>>(dst, deg, E);
    dinv_kernel<<<gdiv(N), 256, 0, stream>>>(deg, dinv, N);
    we_kernel<<<gdiv(E), 256, 0, stream>>>(src, dst, dinv, we, E);

    // weight conversion to bf16 transposed
    convert_wT<<<gdiv(HID * 96), 256, 0, stream>>>(node_w, nodeBt, N_FEAT, 96, HID);
    for (int l = 0; l < 3; ++l)
        for (int k = 0; k < 4; ++k)
            convert_wT<<<gdiv(HID * HID), 256, 0, stream>>>(
                convw[l] + (size_t)k * HID * HID, convBt + (size_t)(l * 4 + k) * HID * HID,
                HID, HID, HID);

    // node encoder: h = relu(x @ W + b)
    gemm_tile32<96><<<N / 32, 128, 0, stream>>>(x, N_FEAT, N_FEAT, nodeBt, acc, N, 1);
    bias_relu<<<gdiv(NH), 256, 0, stream>>>(acc, node_b, h, NH);

    // lattice encoder
    lat_enc<<<G, 128, 0, stream>>>(latf, lat_w, lat_b, lat);

    // TAGConv layers (conv bias cancels under training-mode BN)
    for (int l = 0; l < 3; ++l) {
        const unsigned short* W0 = convBt + (size_t)(l * 4 + 0) * HID * HID;
        const unsigned short* W1 = convBt + (size_t)(l * 4 + 1) * HID * HID;
        const unsigned short* W2 = convBt + (size_t)(l * 4 + 2) * HID * HID;
        const unsigned short* W3 = convBt + (size_t)(l * 4 + 3) * HID * HID;

        gemm_tile32<128><<<N / 32, 128, 0, stream>>>(h, HID, HID, W0, acc, N, 1);

        zero_f32<<<gdiv(NH), 256, 0, stream>>>(pA, NH);
        prop_kernel<<<gdiv((long long)E * 32), 256, 0, stream>>>(h, pA, src, dst, we, E);
        gemm_tile32<128><<<N / 32, 128, 0, stream>>>(pA, HID, HID, W1, acc, N, 0);

        zero_f32<<<gdiv(NH), 256, 0, stream>>>(h, NH);
        prop_kernel<<<gdiv((long long)E * 32), 256, 0, stream>>>(pA, h, src, dst, we, E);
        gemm_tile32<128><<<N / 32, 128, 0, stream>>>(h, HID, HID, W2, acc, N, 0);

        zero_f32<<<gdiv(NH), 256, 0, stream>>>(pA, NH);
        prop_kernel<<<gdiv((long long)E * 32), 256, 0, stream>>>(h, pA, src, dst, we, E);
        gemm_tile32<128><<<N / 32, 128, 0, stream>>>(pA, HID, HID, W3, acc, N, 0);

        // batchnorm (training-mode stats) + relu -> h
        zero_f32<<<1, 256, 0, stream>>>(bnbuf, 256);
        bn_stats<<<gdiv(N), 128, 0, stream>>>(acc, bnbuf, bnbuf + 128, N);
        bn_finalize<<<1, 128, 0, stream>>>(bnbuf, bnbuf + 128, bng[l], bnb[l],
                                           bnbuf + 256, bnbuf + 384, N);
        bn_apply<<<gdiv(NH), 256, 0, stream>>>(acc, bnbuf + 256, bnbuf + 384, h, NH);
    }

    // pooling per graph
    zero_f32<<<gdiv(G * HID), 256, 0, stream>>>(xsum, G * HID);
    zero_f32<<<gdiv(G), 256, 0, stream>>>(counts, G);
    pool_sum<<<gdiv(NH), 256, 0, stream>>>(h, batch, xsum, NH);
    pool_cnt<<<gdiv(N), 256, 0, stream>>>(batch, counts, N);

    // MLP head -> out[G]
    head_kernel<<<G, 128, 0, stream>>>(xsum, counts, lat, fc1w, fc1b, fc2w, fc2b,
                                       fc3w, fc3b, (float*)d_out);
}
